// SpikeGAU_9388798509214
// MI455X (gfx1250) — matrix operand order repaired and checked by reading, not compiled, not run on hardware
//
#include <hip/hip_runtime.h>

// ---------------------------------------------------------------------------
// SpikeGAU on MI455X (gfx1250): all four channel-GEMMs run on
// v_wmma_f32_16x16x32_f16 (binary spike activations are exact in fp16,
// fp32 accumulate). LIF time-recursions are fused into GEMM epilogues.
// LDS tiles are staged with global_load_async_to_lds_b128 (ASYNCcnt),
// double-buffered with one-stage software pipelining.
// ---------------------------------------------------------------------------

typedef _Float16 h8   __attribute__((ext_vector_type(8)));
typedef _Float16 v16h __attribute__((ext_vector_type(16)));
typedef float    v8f  __attribute__((ext_vector_type(8)));
typedef float    f4   __attribute__((ext_vector_type(4)));

union HFrag { v16h v; h8 h[2]; };

typedef __attribute__((address_space(3))) void as3void;

#define T_    4
#define B_    32
#define C_    512
#define HW_   256
#define NCOL  32768      // T*B*HW columns of the big GEMM
#define LDSP  72         // LDS pitch in halves (16B aligned, bank-spread)
#define EPS_  1e-5f

// byte offset of a generic pointer into LDS space (addrspacecast -> AS3)
__device__ __forceinline__ unsigned lds_off(void* p)
{
    return (unsigned)(unsigned long long)(as3void*)p;
}

// async DMA: 16 bytes per lane, global -> LDS, tracked by ASYNCcnt
__device__ __forceinline__ void async_b128(unsigned ldsByteOff, const void* g)
{
    asm volatile("global_load_async_to_lds_b128 %0, %1, off"
                 :: "v"(ldsByteOff), "v"((unsigned long long)g) : "memory");
}

// ---------------------------------------------------------------------------
// Kernel 1: weight conversion fp32 -> fp16.
//   W3  [1536][512]  = stacked gate/q/k weights, row o, c contiguous (A-side)
//   Wpt [512c][512o] = proj weights transposed, o contiguous (B-side)
// ---------------------------------------------------------------------------
__global__ __launch_bounds__(256)
void wconv_kernel(const float* __restrict__ gw, const float* __restrict__ qw,
                  const float* __restrict__ kw, const float* __restrict__ pw,
                  _Float16* __restrict__ W3, _Float16* __restrict__ Wpt)
{
    int tid = blockIdx.x * 256 + threadIdx.x;         // 0 .. 2048*512-1
    if (tid < 1536 * 512) {
        int o = tid >> 9;
        float w;
        if (o < 512)        w = gw[tid];
        else if (o < 1024)  w = qw[tid - 512 * 512];
        else                w = kw[tid - 1024 * 512];
        W3[tid] = (_Float16)w;
    } else {
        int i = tid - 1536 * 512;                     // i = c*512 + o
        int c = i >> 9, o = i & 511;
        Wpt[i] = (_Float16)pw[o * 512 + c];
    }
}

// ---------------------------------------------------------------------------
// Kernel 2: first LIF.  x (T,B,C,H,W) fp32 -> xs [c][col] fp16 (binary).
// col = (t*32+b)*256 + hw.  Reads and writes coalesced along hw.
// ---------------------------------------------------------------------------
__global__ __launch_bounds__(256)
void lif1_kernel(const float* __restrict__ x, _Float16* __restrict__ xs)
{
    int tid = blockIdx.x * 256 + threadIdx.x;         // over B*C*HW = 4194304
    int hw = tid & 255;
    int c  = (tid >> 8) & 511;
    int b  = tid >> 17;
    float v = 0.0f;
    #pragma unroll
    for (int t = 0; t < T_; ++t) {
        float xt = x[(((size_t)(t * 32 + b) * 512 + c) << 8) + hw];
        v += 0.5f * (xt - v);
        float s = (v >= 1.0f) ? 1.0f : 0.0f;
        xs[(size_t)c * NCOL + (((t * 32 + b) << 8) + hw)] = (_Float16)s;
        v = (v >= 1.0f) ? 0.0f : v;
    }
}

// ---------------------------------------------------------------------------
// Kernel 3: fused GEMM1 (gate/q/k) + BN + LIF.
//   D[o,col] = sum_c W3[o,c] * xs[c,col]   (WMMA f16 -> f32)
// Block: 256 thr = 8 waves.  M = 128 output channels (wave = 16 rows).
// Columns: 16 hw x 4 t (one b); accumulator subtile s == time step t, so the
// LIF recursion runs across the 4 accumulators inside each lane.
// B-tile (64K x 64N) staged via async-to-LDS, double-buffered.
// Output spikes stored [col][c] fp16.
// ---------------------------------------------------------------------------
__global__ __launch_bounds__(256)
void gemm1_lif_kernel(const _Float16* __restrict__ W3,
                      const _Float16* __restrict__ xs,
                      _Float16* __restrict__ gate_s,
                      _Float16* __restrict__ q_s,
                      _Float16* __restrict__ k_s,
                      const float* __restrict__ gate_b,
                      const float* __restrict__ gg, const float* __restrict__ gb,
                      const float* __restrict__ gm, const float* __restrict__ gv,
                      const float* __restrict__ qg, const float* __restrict__ qb,
                      const float* __restrict__ qm, const float* __restrict__ qv,
                      const float* __restrict__ kg, const float* __restrict__ kb,
                      const float* __restrict__ km, const float* __restrict__ kv_)
{
    __shared__ __align__(16) _Float16 lds[2][64 * LDSP];

    const int b      = blockIdx.x >> 4;
    const int hwBase = (blockIdx.x & 15) << 4;
    const int oBase  = blockIdx.y * 128;
    const int tid    = threadIdx.x;
    const int wave   = tid >> 5, lane = tid & 31;
    const int nIdx   = lane & 15, halfId = lane >> 4;

    // cooperative fill: 64 K rows x 64 cols, 16 halves (2 async b128)/thread
    const int fk = tid >> 3;                       // 0..31 (and +32)
    const int fc = (tid & 7) << 3;                 // 0..56
    const int ftt = fc >> 4, fhw = fc & 15;
    const size_t fgcol = (size_t)((ftt * B_ + b) * HW_ + hwBase + fhw);

    const int orow = oBase + wave * 16 + nIdx;     // A-matrix row (lane = M)
    const v8f zero = {0.f, 0.f, 0.f, 0.f, 0.f, 0.f, 0.f, 0.f};
    v8f acc[4] = {zero, zero, zero, zero};

    auto fill = [&](int stage) {
        int p = stage & 1;
        int kk = stage << 6;
        const _Float16* g0 = xs + (size_t)(kk + fk) * NCOL + fgcol;
        async_b128(lds_off(&lds[p][fk * LDSP + fc]), g0);
        async_b128(lds_off(&lds[p][(fk + 32) * LDSP + fc]), g0 + (size_t)32 * NCOL);
    };

    fill(0);
    for (int i = 0; i < 8; ++i) {
        if (i < 7) {
            fill(i + 1);
            asm volatile("s_wait_asynccnt 0x2" ::: "memory");
        } else {
            asm volatile("s_wait_asynccnt 0x0" ::: "memory");
        }
        __syncthreads();

        const _Float16* buf = lds[i & 1];
        const int kk = i << 6;
        #pragma unroll
        for (int kh = 0; kh < 2; ++kh) {
            HFrag A;
            const size_t wrow = (size_t)orow * C_ + kk + kh * 32;
            A.h[0] = *(const h8*)&W3[wrow + 8 * halfId];
            A.h[1] = *(const h8*)&W3[wrow + 16 + 8 * halfId];
            #pragma unroll
            for (int s = 0; s < 4; ++s) {          // B lane = N col
                HFrag Bf;
                const _Float16* bp = &buf[(kh * 32 + 8 * halfId) * LDSP + s * 16 + nIdx];
                #pragma unroll
                for (int e = 0; e < 8; ++e) {
                    Bf.h[0][e] = bp[e * LDSP];
                    Bf.h[1][e] = bp[(16 + e) * LDSP];
                }
                acc[s] = __builtin_amdgcn_wmma_f32_16x16x32_f16(
                    false, A.v, false, Bf.v, (short)0, acc[s], false, false);
            }
        }
        __syncthreads();
    }

    // ---- epilogue: BN + LIF over the 4 t-accumulators --------------------
    const int pathSel = blockIdx.y >> 2;           // 0=gate 1=q 2=k
    const float *Pg, *Pb, *Pm, *Pv;
    if (pathSel == 0)      { Pg = gg; Pb = gb; Pm = gm; Pv = gv; }
    else if (pathSel == 1) { Pg = qg; Pb = qb; Pm = qm; Pv = qv; }
    else                   { Pg = kg; Pb = kb; Pm = km; Pv = kv_; }

    const int cBase = ((blockIdx.y & 3) << 7) + wave * 16 + halfId * 8;
    float scale[8], shift[8];
    #pragma unroll
    for (int r = 0; r < 8; ++r) {
        int c = cBase + r;
        float rs = rsqrtf(Pv[c] + EPS_);
        scale[r] = Pg[c] * rs;
        shift[r] = Pb[c] - Pg[c] * Pm[c] * rs;
        if (pathSel == 0) shift[r] += gate_b[c] * scale[r];
    }

    h8 sp[4];
    #pragma unroll
    for (int r = 0; r < 8; ++r) {
        float v = 0.0f;
        #pragma unroll
        for (int s = 0; s < 4; ++s) {              // s == time step t
            float z = acc[s][r] * scale[r] + shift[r];
            v += 0.5f * (z - v);
            float spk = (v >= 1.0f) ? 1.0f : 0.0f;
            sp[s][r] = (_Float16)spk;
            v = (v >= 1.0f) ? 0.0f : v;
        }
    }

    _Float16* dst = (pathSel == 0) ? gate_s : ((pathSel == 1) ? q_s : k_s);
    #pragma unroll
    for (int s = 0; s < 4; ++s) {
        size_t gcol = (size_t)((s * B_ + b) * HW_ + hwBase + nIdx);
        *(h8*)&dst[gcol * C_ + cBase] = sp[s];
    }
}

// ---------------------------------------------------------------------------
// Kernel 4: kv = lif( sum_hw k_s , v_th=0.5 ).  k binary => k*v == k.
// One thread per (b,c); coalesced along c.  Output kv [t][b][c] fp16 binary.
// ---------------------------------------------------------------------------
__global__ __launch_bounds__(256)
void kv_kernel(const _Float16* __restrict__ ks, _Float16* __restrict__ kvh)
{
    int tid = blockIdx.x * 256 + threadIdx.x;      // 16384 = B*C
    int b = tid >> 9, c = tid & 511;
    float v = 0.0f;
    for (int t = 0; t < T_; ++t) {
        float sum = 0.0f;
        int colBase = (t * 32 + b) << 8;
        for (int hw = 0; hw < HW_; ++hw)
            sum += (float)ks[(size_t)(colBase + hw) * C_ + c];
        v += 0.5f * (sum - v);
        float s = (v >= 0.5f) ? 1.0f : 0.0f;
        kvh[(size_t)(t * 32 + b) * C_ + c] = (_Float16)s;
        v = (v >= 0.5f) ? 0.0f : v;
    }
}

// ---------------------------------------------------------------------------
// Kernel 5: second output v = to_heads(k spikes), fp32, at d_out + 16M.
// ---------------------------------------------------------------------------
__global__ __launch_bounds__(256)
void vout_kernel(const _Float16* __restrict__ ks, float* __restrict__ out)
{
    size_t tid = (size_t)blockIdx.x * 256 + threadIdx.x;   // 16777216
    int ch = (int)(tid & 63);
    int n  = (int)((tid >> 6) & 255);
    int hd = (int)((tid >> 14) & 7);
    int tb = (int)(tid >> 17);                     // t*32 + b
    int c  = hd * 64 + ch;
    out[16777216u + tid] = (float)ks[((size_t)tb * HW_ + n) * C_ + c];
}

// ---------------------------------------------------------------------------
// Kernel 6: proj GEMM (roles flipped: A = activation tile, B = Wpt in LDS).
//   a[c,col] = q_s * gate_s * kv  formed in registers during fragment load.
//   out[o,col] = BN(proj(a)) + identity x.
// Block: 8 waves, M = 128 cols (one (t,b), half an image), N = 64 channels.
// Wpt tile (64K x 64N) staged via async-to-LDS, double-buffered.
// ---------------------------------------------------------------------------
__global__ __launch_bounds__(256)
void gemm2_kernel(const _Float16* __restrict__ Wpt,
                  const _Float16* __restrict__ q_s,
                  const _Float16* __restrict__ gate_s,
                  const _Float16* __restrict__ kvh,
                  const float* __restrict__ x,
                  const float* __restrict__ proj_b,
                  const float* __restrict__ pg, const float* __restrict__ pb,
                  const float* __restrict__ pm, const float* __restrict__ pv,
                  float* __restrict__ out)
{
    __shared__ __align__(16) _Float16 lds[2][64 * LDSP];

    const int tb    = blockIdx.x >> 1;             // t*32 + b
    const int hwB   = (blockIdx.x & 1) << 7;       // 0 or 128
    const int oBase = blockIdx.y << 6;             // 64-channel tile
    const int tid   = threadIdx.x;
    const int wave  = tid >> 5, lane = tid & 31;
    const int nIdx  = lane & 15, halfId = lane >> 4;

    const int fk = tid >> 3;                       // 0..31 (and +32)
    const int fo = (tid & 7) << 3;

    const size_t colA  = (size_t)tb * HW_ + hwB + wave * 16 + nIdx;  // A row
    const size_t kvRow = (size_t)tb * C_;

    const v8f zero = {0.f, 0.f, 0.f, 0.f, 0.f, 0.f, 0.f, 0.f};
    v8f acc[4] = {zero, zero, zero, zero};

    auto fill = [&](int stage) {
        int p = stage & 1;
        int kk = stage << 6;
        const _Float16* g0 = Wpt + (size_t)(kk + fk) * C_ + oBase + fo;
        async_b128(lds_off(&lds[p][fk * LDSP + fo]), g0);
        async_b128(lds_off(&lds[p][(fk + 32) * LDSP + fo]), g0 + (size_t)32 * C_);
    };

    fill(0);
    for (int i = 0; i < 8; ++i) {
        if (i < 7) {
            fill(i + 1);
            asm volatile("s_wait_asynccnt 0x2" ::: "memory");
        } else {
            asm volatile("s_wait_asynccnt 0x0" ::: "memory");
        }
        __syncthreads();

        const _Float16* buf = lds[i & 1];
        const int kk = i << 6;
        #pragma unroll
        for (int kh = 0; kh < 2; ++kh) {
            HFrag A;
            {
                size_t base0 = colA * C_ + kk + kh * 32 + 8 * halfId;
                h8 qa = *(const h8*)&q_s[base0];
                h8 ga = *(const h8*)&gate_s[base0];
                h8 va = *(const h8*)&kvh[kvRow + kk + kh * 32 + 8 * halfId];
                A.h[0] = qa * ga * va;
                h8 qc = *(const h8*)&q_s[base0 + 16];
                h8 gc = *(const h8*)&gate_s[base0 + 16];
                h8 vc = *(const h8*)&kvh[kvRow + kk + kh * 32 + 16 + 8 * halfId];
                A.h[1] = qc * gc * vc;
            }
            #pragma unroll
            for (int s = 0; s < 4; ++s) {
                HFrag Bf;
                const _Float16* bp = &buf[(kh * 32 + 8 * halfId) * LDSP + s * 16 + nIdx];
                #pragma unroll
                for (int e = 0; e < 8; ++e) {
                    Bf.h[0][e] = bp[e * LDSP];
                    Bf.h[1][e] = bp[(16 + e) * LDSP];
                }
                acc[s] = __builtin_amdgcn_wmma_f32_16x16x32_f16(
                    false, A.v, false, Bf.v, (short)0, acc[s], false, false);
            }
        }
        __syncthreads();
    }

    // ---- epilogue: BN (+bias) + identity, float4 stores ------------------
    const int hw0 = hwB + wave * 16 + halfId * 8;  // 8 contiguous hw (r)
    #pragma unroll
    for (int s = 0; s < 4; ++s) {
        int o = oBase + s * 16 + nIdx;
        float rs = rsqrtf(pv[o] + EPS_);
        float sc = pg[o] * rs;
        float sh = pb[o] - pg[o] * pm[o] * rs + proj_b[o] * sc;
        size_t idx = ((size_t)tb * C_ + o) * HW_ + hw0;
        const f4* xp = (const f4*)&x[idx];
        f4 x0 = xp[0], x1 = xp[1];
        f4 r0, r1;
        #pragma unroll
        for (int j = 0; j < 4; ++j) {
            r0[j] = acc[s][j]     * sc + sh + x0[j];
            r1[j] = acc[s][4 + j] * sc + sh + x1[j];
        }
        f4* op = (f4*)&out[idx];
        op[0] = r0;
        op[1] = r1;
    }
}

// ---------------------------------------------------------------------------
// Launch. Workspace layout (bytes), total ~136.5 MB:
//   W3   @ 0          1,572,864
//   Wpt  @ 1,572,864    524,288
//   xs   @ 2,097,152 33,554,432
//   gate @ 35,651,584 33,554,432
//   q    @ 69,206,016 33,554,432
//   k    @102,760,448 33,554,432
//   kv   @136,314,880    131,072
// ---------------------------------------------------------------------------
extern "C" void kernel_launch(void* const* d_in, const int* in_sizes, int n_in,
                              void* d_out, int out_size, void* d_ws, size_t ws_size,
                              hipStream_t stream)
{
    const float* x      = (const float*)d_in[0];
    const float* gate_w = (const float*)d_in[1];
    const float* gate_b = (const float*)d_in[2];
    const float* q_w    = (const float*)d_in[3];
    const float* k_w    = (const float*)d_in[4];
    const float* proj_w = (const float*)d_in[5];
    const float* proj_b = (const float*)d_in[6];
    const float* gg = (const float*)d_in[7],  *gb = (const float*)d_in[8];
    const float* gm = (const float*)d_in[9],  *gv = (const float*)d_in[10];
    const float* qg = (const float*)d_in[11], *qb = (const float*)d_in[12];
    const float* qm = (const float*)d_in[13], *qv = (const float*)d_in[14];
    const float* kg = (const float*)d_in[15], *kb = (const float*)d_in[16];
    const float* km = (const float*)d_in[17], *kv = (const float*)d_in[18];
    const float* pg = (const float*)d_in[19], *pb = (const float*)d_in[20];
    const float* pm = (const float*)d_in[21], *pv = (const float*)d_in[22];

    char* ws = (char*)d_ws;
    _Float16* W3   = (_Float16*)(ws);
    _Float16* Wpt  = (_Float16*)(ws + 1572864);
    _Float16* xs   = (_Float16*)(ws + 2097152);
    _Float16* gsb  = (_Float16*)(ws + 35651584);
    _Float16* qsb  = (_Float16*)(ws + 69206016);
    _Float16* ksb  = (_Float16*)(ws + 102760448);
    _Float16* kvh  = (_Float16*)(ws + 136314880);
    float* out = (float*)d_out;

    wconv_kernel<<<4096, 256, 0, stream>>>(gate_w, q_w, k_w, proj_w, W3, Wpt);
    lif1_kernel<<<16384, 256, 0, stream>>>(x, xs);
    gemm1_lif_kernel<<<dim3(512, 12), 256, 0, stream>>>(
        W3, xs, gsb, qsb, ksb, gate_b,
        gg, gb, gm, gv, qg, qb, qm, qv, kg, kb, km, kv);
    kv_kernel<<<64, 256, 0, stream>>>(ksb, kvh);
    vout_kernel<<<65536, 256, 0, stream>>>(ksb, out);
    gemm2_kernel<<<dim3(256, 8), 256, 0, stream>>>(
        Wpt, qsb, gsb, kvh, x, proj_b, pg, pb, pm, pv, out);
}
